// GNNClassifier_76519137345589
// MI455X (gfx1250) — matrix operand order repaired and checked by reading, not compiled, not run on hardware
//
#include <hip/hip_runtime.h>
#include <hip/hip_bf16.h>
#include <math.h>

typedef __attribute__((ext_vector_type(16))) __bf16 v16bf;
typedef __attribute__((ext_vector_type(8)))  __bf16 v8bf;
typedef __attribute__((ext_vector_type(4)))  __bf16 v4bf;
typedef __attribute__((ext_vector_type(8)))  float  v8f;

constexpr int   NN    = 50000;
constexpr int   EE    = 600000;
constexpr int   NGRAPH= 16;
constexpr int   NPOOL = 12496;
constexpr int   NPER  = 781;
constexpr int   EPOOL = 149952;
constexpr int   D1 = 32, D2 = 64, D3 = 128;
constexpr int   NCLS  = 100;
constexpr int   KC    = 27;     // 3^3 spline kernels
constexpr float EPSBN = 1e-5f;

// ---------------- small utility kernels ----------------

__global__ __launch_bounds__(256) void k_zero(float* p, int n) {
  int i = blockIdx.x * 256 + threadIdx.x;
  if (i < n) p[i] = 0.f;
}

__global__ __launch_bounds__(256) void k_emb(const int* __restrict__ tok,
                                             const float* __restrict__ emb,
                                             float* __restrict__ x, int n) {
  int i = blockIdx.x * 256 + threadIdx.x;
  if (i >= n * D1) return;
  int node = i >> 5, d = i & 31;
  x[i] = emb[tok[node] * D1 + d];
}

__global__ __launch_bounds__(256) void k_count(const int* __restrict__ dst,
                                               float* cnt, int e) {
  int i = blockIdx.x * 256 + threadIdx.x;
  if (i < e) atomicAdd(&cnt[dst[i]], 1.f);
}

__global__ __launch_bounds__(256) void k_invcnt(float* cnt, int n) {
  int i = blockIdx.x * 256 + threadIdx.x;
  if (i < n) cnt[i] = 1.f / fmaxf(cnt[i], 1.f);
}

// pack [W(27*di,do); R(di,do)] -> bf16 TRANSPOSED [DO][K] (column-major in k)
__global__ __launch_bounds__(256) void k_prepw(const float* __restrict__ W,
                                               const float* __restrict__ R,
                                               __bf16* __restrict__ WbT,
                                               int wcnt, int rcnt, int DO) {
  int i = blockIdx.x * 256 + threadIdx.x;
  if (i >= wcnt + rcnt) return;
  float v = (i < wcnt) ? W[i] : R[i - wcnt];
  int K = (wcnt + rcnt) / DO;
  int k = i / DO, o = i % DO;
  WbT[(size_t)o * K + k] = (__bf16)v;
}

// S[n, 0:27*di] = 0 ; S[n, 27*di:28*di] = x[n, :]   (x*R rides in the GEMM tail)
__global__ __launch_bounds__(256) void k_initS(float* __restrict__ S,
                                               const float* __restrict__ x,
                                               int n, int di, int K) {
  long long i = (long long)blockIdx.x * 256 + threadIdx.x;
  long long tot = (long long)n * K;
  if (i >= tot) return;
  int col = (int)(i % K);
  int node = (int)(i / K);
  S[i] = (col >= KC * di) ? x[(size_t)node * di + (col - KC * di)] : 0.f;
}

// ---------------- edge scatter: wave32 per edge, lanes cover 32 consecutive features ----------------

__global__ __launch_bounds__(256) void k_scatter(
    const int* __restrict__ eidx, const float* __restrict__ posv,
    const float* __restrict__ x, const float* __restrict__ icnt,
    float* __restrict__ S, int e, int di, int K, float inv2max) {
  int w    = (blockIdx.x * 256 + threadIdx.x) >> 5;   // one wave per edge
  int lane = threadIdx.x & 31;
  if (w >= e) return;
  int src = eidx[w];
  int dst = eidx[e + w];
  int lo[3]; float fr[3];
#pragma unroll
  for (int d = 0; d < 3; ++d) {
    float c  = posv[src * 3 + d] - posv[dst * 3 + d];
    float ps = fminf(fmaxf(c * inv2max + 0.5f, 0.f), 1.f);   // Cartesian(norm=True)
    float v  = fminf(ps * 2.f, 2.f - 1e-6f);                 // pseudo*(KS-1), clipped
    float l  = floorf(v);
    lo[d] = (int)l; fr[d] = v - l;
  }
  float ic = icnt[dst];
  const float* xs = x + (size_t)src * di;
  float* Sd = S + (size_t)dst * K;
  const int nf = di >> 5;                  // 1, 2 or 4 features per lane
  float xv[4];
#pragma unroll 4
  for (int t = 0; t < nf; ++t) xv[t] = xs[lane + 32 * t];
#pragma unroll
  for (int bits = 0; bits < 8; ++bits) {
    int b0 = bits & 1, b1 = (bits >> 1) & 1, b2 = (bits >> 2) & 1;
    int idx = (lo[0] + b0) + 3 * (lo[1] + b1) + 9 * (lo[2] + b2);
    float bas = ic * (b0 ? fr[0] : 1.f - fr[0])
                   * (b1 ? fr[1] : 1.f - fr[1])
                   * (b2 ? fr[2] : 1.f - fr[2]);
    float* t0 = Sd + idx * di;
#pragma unroll 4
    for (int t = 0; t < nf; ++t)
      atomicAdd(&t0[lane + 32 * t], bas * xv[t]);             // coalesced atomic burst
  }
}

// ---------------- WMMA GEMM: y[M,DO] = elu(S[M,K] @ W[K,DO] + bias), + BN column stats ----------------
// A staged in LDS in fragment order: aF[row][half][e], holds k = 16*half + e (contiguous, same order as B).
// B staged from pre-transposed WbT[DO][K]: lane fragment is k = 16*half .. +15 (contiguous).
// Each wave owns TWO m-tiles: 2 WMMAs share one B fragment and one barrier pair per k-chunk.

template <int DO>
__global__ __launch_bounds__(256) void k_gemm(
    const float* __restrict__ S, const int K, const __bf16* __restrict__ WbT,
    const float* __restrict__ bias, float* __restrict__ y,
    float* __restrict__ stats, const int M) {
  constexpr int NT  = DO / 16;                // n-tiles (4 or 8)
  constexpr int MG  = 8 / NT;                 // m-groups of waves (2 or 1)
  constexpr int BM  = MG * 32;                // nodes per block: 64 or 32
  constexpr int AV  = (BM * 32) / 256;        // f32 per thread for A: 8 or 4
  __shared__ alignas(32) __bf16 aF[BM][2][16];
  __shared__ alignas(32) __bf16 bT[DO][32];
  __shared__ float cs1[DO], cs2[DO];

  const int tid  = threadIdx.x;
  const int wave = tid >> 5, lane = tid & 31;
  const int mgrp = wave / NT, ntile = wave % NT;
  const int nodeBase = blockIdx.x * BM;
  const int half = lane >> 4;
  const int l15  = lane & 15;
  const int ncol = ntile * 16 + l15;          // B/D: lane holds column N = lane&15

  if (tid < DO) { cs1[tid] = 0.f; cs2[tid] = 0.f; }

  // --- compile-time staging coordinates (c -> (half, e) inversion of A layout) ---
  const int ar = (tid * AV) >> 5;
  const int ac = (tid * AV) & 31;             // multiple of AV; stays in one (h, e-run)
  const int ah = ac >> 4;
  const int ae = ac & 15;
  const int anode = nodeBase + ar;
  const float* aSrc = S + (size_t)anode * K + ac;

  constexpr int BCH = (32 * DO) / 256 / 8;    // 8-wide bf16 chunks per thread: 1 or 2
  v8f acc0 = {}, acc1 = {};

  for (int kc = 0; kc < K; kc += 32) {
    __syncthreads();
    // ---- stage A: vector loads, packed f32->bf16, one vector LDS store ----
    if constexpr (AV == 8) {
      float4 u = make_float4(0.f, 0.f, 0.f, 0.f), w = u;
      if (anode < M) {
        u = *reinterpret_cast<const float4*>(aSrc + kc);
        w = *reinterpret_cast<const float4*>(aSrc + kc + 4);
      }
      v8bf p;
      p[0] = (__bf16)u.x; p[1] = (__bf16)u.y; p[2] = (__bf16)u.z; p[3] = (__bf16)u.w;
      p[4] = (__bf16)w.x; p[5] = (__bf16)w.y; p[6] = (__bf16)w.z; p[7] = (__bf16)w.w;
      *reinterpret_cast<v8bf*>(&aF[ar][ah][ae]) = p;
    } else {
      float4 u = (anode < M) ? *reinterpret_cast<const float4*>(aSrc + kc)
                             : make_float4(0.f, 0.f, 0.f, 0.f);
      v4bf p; p[0] = (__bf16)u.x; p[1] = (__bf16)u.y;
              p[2] = (__bf16)u.z; p[3] = (__bf16)u.w;
      *reinterpret_cast<v4bf*>(&aF[ar][ah][ae]) = p;
    }
    // ---- stage B: 16B vector copies from transposed weights ----
#pragma unroll
    for (int ch = 0; ch < BCH; ++ch) {
      int li  = (tid * BCH + ch) * 8;
      int col = li >> 5;
      int kk  = li & 31;
      *reinterpret_cast<v8bf*>(&bT[col][kk]) =
          *reinterpret_cast<const v8bf*>(&WbT[(size_t)col * K + kc + kk]);
    }
    if (kc + 32 < K)
      __builtin_prefetch(aSrc + kc + 32, 0, 1);   // global_prefetch_b8 for next A chunk
    __syncthreads();

    // ---- fragments: contiguous 32B LDS reads; B reused by both m-tiles ----
    const v16bf b  = *reinterpret_cast<const v16bf*>(&bT[ncol][16 * half]);
    const v16bf a0 = *reinterpret_cast<const v16bf*>(&aF[(mgrp * 2 + 0) * 16 + l15][half][0]);
    const v16bf a1 = *reinterpret_cast<const v16bf*>(&aF[(mgrp * 2 + 1) * 16 + l15][half][0]);
    acc0 = __builtin_amdgcn_wmma_f32_16x16x32_bf16(false, a0, false, b,
                                                   (short)0, acc0, false, false);
    acc1 = __builtin_amdgcn_wmma_f32_16x16x32_bf16(false, a1, false, b,
                                                   (short)0, acc1, false, false);
  }
  __syncthreads();

  // epilogue: D layout lane l, vgpr r -> (M = r + 8*(l>>4), N = l&15)
#pragma unroll
  for (int s = 0; s < 2; ++s) {
    const v8f& acc = s ? acc1 : acc0;
#pragma unroll
    for (int r = 0; r < 8; ++r) {
      int m = (mgrp * 2 + s) * 16 + r + 8 * half;
      int node = nodeBase + m;
      if (node < M) {
        float v = acc[r] + bias[ncol];
        float e = (v > 0.f) ? v : (__expf(v) - 1.f);   // ELU
        y[(size_t)node * DO + ncol] = e;
        atomicAdd(&cs1[ncol], e);
        atomicAdd(&cs2[ncol], e * e);
      }
    }
  }
  __syncthreads();
  if (tid < DO) {
    atomicAdd(&stats[tid], cs1[tid]);
    atomicAdd(&stats[DO + tid], cs2[tid]);
  }
}

// ---------------- BN finalize / apply ----------------

__global__ void k_bnfin(const float* __restrict__ stats,
                        const float* __restrict__ g, const float* __restrict__ be,
                        float* __restrict__ ss, int M, int DO) {
  int o = threadIdx.x;
  if (o >= DO) return;
  float mean = stats[o] / (float)M;
  float var  = stats[DO + o] / (float)M - mean * mean;
  float sc   = g[o] * rsqrtf(var + EPSBN);
  ss[o]      = sc;
  ss[DO + o] = be[o] - mean * sc;
}

__global__ __launch_bounds__(256) void k_bnapply(const float* __restrict__ y,
                                                 const float* __restrict__ ss,
                                                 const float* __restrict__ skip,
                                                 float* __restrict__ xout,
                                                 int M, int DO) {
  long long i = (long long)blockIdx.x * 256 + threadIdx.x;
  if (i >= (long long)M * DO) return;
  int o = (int)(i % DO);
  float v = y[i] * ss[o] + ss[DO + o];
  if (skip) v += skip[i];
  xout[i] = v;
}

// ---------------- pooling / head ----------------

__global__ __launch_bounds__(256) void k_gather(const float* __restrict__ x,
                                                const int* __restrict__ idxv,
                                                float* __restrict__ out,
                                                int n, int dim) {
  int i = blockIdx.x * 256 + threadIdx.x;
  if (i >= n * dim) return;
  int j = i / dim, d = i % dim;
  out[i] = x[(size_t)idxv[j] * dim + d];
}

__global__ void k_segmax(const float* __restrict__ x, float* __restrict__ g,
                         int per, int DO) {
  int gi = blockIdx.x, o = threadIdx.x;
  if (o >= DO) return;
  float m = -3.4e38f;
  for (int j = 0; j < per; ++j)
    m = fmaxf(m, x[((size_t)(gi * per + j)) * DO + o]);
  g[gi * DO + o] = m;
}

__global__ void k_fc(const float* __restrict__ gm, const float* __restrict__ Wf,
                     const float* __restrict__ bf, float* __restrict__ out) {
  int gi = blockIdx.x, t = threadIdx.x;
  if (t >= NCLS) return;
  float s = bf[t];
  for (int i = 0; i < D3; ++i) s += gm[gi * D3 + i] * Wf[i * NCLS + t];
  out[gi * NCLS + t] = s;
}

// ---------------- host orchestration ----------------

extern "C" void kernel_launch(void* const* d_in, const int* in_sizes, int n_in,
                              void* d_out, int out_size, void* d_ws, size_t ws_size,
                              hipStream_t stream) {
  (void)in_sizes; (void)n_in; (void)out_size; (void)ws_size;
  const int*   x_tok = (const int*)  d_in[0];
  const float* pos   = (const float*)d_in[1];
  const int*   ei    = (const int*)  d_in[3];
  const int*   pidx  = (const int*)  d_in[4];
  const int*   pei   = (const int*)  d_in[5];
  const float* emb   = (const float*)d_in[6];
  auto Wp  = [&](int li) { return (const float*)d_in[7 + (li - 1) * 5 + 0]; };
  auto Rp  = [&](int li) { return (const float*)d_in[7 + (li - 1) * 5 + 1]; };
  auto bp  = [&](int li) { return (const float*)d_in[7 + (li - 1) * 5 + 2]; };
  auto gp  = [&](int li) { return (const float*)d_in[7 + (li - 1) * 5 + 3]; };
  auto bep = [&](int li) { return (const float*)d_in[7 + (li - 1) * 5 + 4]; };
  const float* fcW = (const float*)d_in[42];
  const float* fcb = (const float*)d_in[43];
  float* out = (float*)d_out;

  char* wsp = (char*)d_ws;
  auto alloc = [&](size_t bytes) -> void* {
    void* p = wsp; wsp += (bytes + 255) & ~(size_t)255; return p;
  };
  float*  S    = (float*) alloc((size_t)NN * 28 * D2 * sizeof(float)); // 89.6M f32; pooled fits too
  float*  y    = (float*) alloc((size_t)NN * D3 * sizeof(float));
  float*  xA   = (float*) alloc((size_t)NN * D3 * sizeof(float));
  float*  xB   = (float*) alloc((size_t)NN * D3 * sizeof(float));
  float*  xSC  = (float*) alloc((size_t)NN * D3 * sizeof(float));
  float*  xp   = (float*) alloc((size_t)NPOOL * D3 * sizeof(float));
  float*  posp = (float*) alloc((size_t)NPOOL * 3 * sizeof(float));
  float*  cntN = (float*) alloc((size_t)NN * sizeof(float));
  float*  cntP = (float*) alloc((size_t)NPOOL * sizeof(float));
  float*  stats= (float*) alloc(256 * sizeof(float));
  float*  ss   = (float*) alloc(256 * sizeof(float));
  __bf16* WbT  = (__bf16*)alloc((size_t)28 * D3 * D3 * sizeof(__bf16));
  float*  gm   = (float*) alloc((size_t)NGRAPH * D3 * sizeof(float));

  auto cdiv = [](long long a, long long b) { return (int)((a + b - 1) / b); };

  // per-graph degree -> 1/max(cnt,1) (shared by all layers on that graph)
  k_zero  <<<cdiv(NN, 256),    256, 0, stream>>>(cntN, NN);
  k_count <<<cdiv(EE, 256),    256, 0, stream>>>(ei + EE, cntN, EE);
  k_invcnt<<<cdiv(NN, 256),    256, 0, stream>>>(cntN, NN);
  k_zero  <<<cdiv(NPOOL, 256), 256, 0, stream>>>(cntP, NPOOL);
  k_count <<<cdiv(EPOOL, 256), 256, 0, stream>>>(pei + EPOOL, cntP, EPOOL);
  k_invcnt<<<cdiv(NPOOL, 256), 256, 0, stream>>>(cntP, NPOOL);

  k_emb<<<cdiv((long long)NN * D1, 256), 256, 0, stream>>>(x_tok, emb, xA, NN);

  auto layer = [&](const float* xin, int di, int DO, float* xout, const float* skip,
                   int n, int e, const int* eidx, const float* posv,
                   const float* icnt, int li, float inv2max) {
    int K = 28 * di;
    k_prepw<<<cdiv((long long)K * DO, 256), 256, 0, stream>>>(
        Wp(li), Rp(li), WbT, KC * di * DO, di * DO, DO);
    k_initS<<<cdiv((long long)n * K, 256), 256, 0, stream>>>(S, xin, n, di, K);
    k_scatter<<<cdiv((long long)e * 32, 256), 256, 0, stream>>>(
        eidx, posv, xin, icnt, S, e, di, K, inv2max);
    k_zero<<<1, 256, 0, stream>>>(stats, 2 * DO);
    if (DO == 64)
      k_gemm<64><<<cdiv(n, 64), 256, 0, stream>>>(S, K, WbT, bp(li), y, stats, n);
    else
      k_gemm<128><<<cdiv(n, 32), 256, 0, stream>>>(S, K, WbT, bp(li), y, stats, n);
    k_bnfin<<<1, DO, 0, stream>>>(stats, gp(li), bep(li), ss, n, DO);
    k_bnapply<<<cdiv((long long)n * DO, 256), 256, 0, stream>>>(y, ss, skip, xout, n, DO);
  };

  const float INV2_NODE = 5.0f;  // 1/(2*0.1)
  const float INV2_POOL = 2.5f;  // 1/(2*0.2)

  layer(xA,  D1, D2, xB,  nullptr, NN, EE, ei, pos, cntN, 1, INV2_NODE);
  layer(xB,  D2, D2, xSC, nullptr, NN, EE, ei, pos, cntN, 2, INV2_NODE);
  layer(xSC, D2, D2, xA,  nullptr, NN, EE, ei, pos, cntN, 3, INV2_NODE);
  layer(xA,  D2, D2, xB,  xSC,     NN, EE, ei, pos, cntN, 4, INV2_NODE);
  layer(xB,  D2, D3, xA,  nullptr, NN, EE, ei, pos, cntN, 5, INV2_NODE);

  k_gather<<<cdiv((long long)NPOOL * D3, 256), 256, 0, stream>>>(xA, pidx, xp, NPOOL, D3);
  k_gather<<<cdiv((long long)NPOOL * 3, 256), 256, 0, stream>>>(pos, pidx, posp, NPOOL, 3);

  layer(xp, D3, D3, xB, nullptr, NPOOL, EPOOL, pei, posp, cntP, 6, INV2_POOL);
  layer(xB, D3, D3, xA, xp,      NPOOL, EPOOL, pei, posp, cntP, 7, INV2_POOL);

  k_segmax<<<NGRAPH, D3, 0, stream>>>(xA, gm, NPER, D3);
  k_fc<<<NGRAPH, 128, 0, stream>>>(gm, fcW, fcb, out);
}
